// ProbabilisticHebbian_56221121904852
// MI455X (gfx1250) — matrix operand order repaired and checked by reading, not compiled, not run on hardware
//
#include <hip/hip_runtime.h>
#include <hip/hip_bf16.h>

typedef __attribute__((ext_vector_type(16))) _Float16 v16h;
typedef __attribute__((ext_vector_type(8)))  _Float16 v8h;
typedef __attribute__((ext_vector_type(8)))  float    v8f;
typedef __attribute__((ext_vector_type(4)))  float    v4f;

#define LR      0.005f
#define WD      0.0001f
#define BATCH   128
#define DIM     1024      // POST == PRE == 1024
#define TILE    64        // per-block output tile (64x64) -> 16x16 grid = 256 blocks
#define LDS_K   136       // 128 + 8 halfs pad; row stride 272B (16B aligned)

// dw[p][q] = LR * ( (1/B) * sum_k post[k][p]*pre[k][q]  -  WD * W[p][q] )
__global__ __launch_bounds__(128) void hebbian_wmma_kernel(
    const float* __restrict__ pre,    // (128, 1024) row-major: pre[k*DIM + q]
    const float* __restrict__ post,   // (128, 1024) row-major: post[k*DIM + p]
    const float* __restrict__ weights,// (1024, 1024)
    float* __restrict__ out)          // (1024, 1024)
{
    // Transposed f16 staging: s*[output_dim_local][k]
    __shared__ __align__(16) _Float16 sA[TILE][LDS_K]; // postT: sA[m][k]
    __shared__ __align__(16) _Float16 sB[TILE][LDS_K]; // preT:  sB[n][k]

    const int tid    = threadIdx.x;          // 0..127 (4 wave32s)
    const int blockM = blockIdx.y * TILE;
    const int blockN = blockIdx.x * TILE;

    // Warm the epilogue's weights tile (64 rows x 256B = 128 x 128B lines;
    // one speculative prefetch per thread) -> global_prefetch_b8.
    __builtin_prefetch(weights + (blockM + (tid & 63)) * DIM + blockN + (tid >> 6) * 32,
                       0, 3);

    // ---- Stage both 128(k) x 64 tiles, converting f32 -> f16, transposing.
    // 2048 float4 per tile; 16 per thread. 16 consecutive lanes cover one
    // contiguous 256B k-row segment -> coalesced global_load_b128.
#pragma unroll
    for (int i = 0; i < 16; ++i) {
        const int f  = i * 128 + tid;       // float4 index within 128x64 tile
        const int k  = f >> 4;              // 0..127 (batch index)
        const int c4 = (f & 15) << 2;       // 0,4,...,60 (output-dim local)

        v4f va = *(const v4f*)(post + k * DIM + blockM + c4);
        sA[c4 + 0][k] = (_Float16)va[0];
        sA[c4 + 1][k] = (_Float16)va[1];
        sA[c4 + 2][k] = (_Float16)va[2];
        sA[c4 + 3][k] = (_Float16)va[3];

        v4f vb = *(const v4f*)(pre + k * DIM + blockN + c4);
        sB[c4 + 0][k] = (_Float16)vb[0];
        sB[c4 + 1][k] = (_Float16)vb[1];
        sB[c4 + 2][k] = (_Float16)vb[2];
        sB[c4 + 3][k] = (_Float16)vb[3];
    }
    __syncthreads();

    // ---- Compute: wave w owns output rows [w*16, w*16+16) of the block,
    // across all 4 column tiles. 4 K-chunks of 32 -> 16 WMMAs per wave.
    const int wave = tid >> 5;
    const int lane = tid & 31;
    const int h    = lane >> 4;   // lane half selects K sub-range
    const int r    = lane & 15;

    v8f acc[4];
#pragma unroll
    for (int n = 0; n < 4; ++n) acc[n] = (v8f){};

    const int mloc = wave * 16 + r;   // A-matrix row for this lane

#pragma unroll
    for (int kc = 0; kc < 4; ++kc) {
        const int kk = kc * 32;

        // A fragment (16x32 f16): VGPR0-3 = K hb..hb+7, VGPR4-7 = K hb+16..hb+23,
        // hb = kk + h*8.  Two contiguous ds_load_b128.
        const _Float16* ap = &sA[mloc][kk + h * 8];
        v8h alo = *(const v8h*)ap;
        v8h ahi = *(const v8h*)(ap + 16);
        v16h a = __builtin_shufflevector(alo, ahi,
                 0,1,2,3,4,5,6,7,8,9,10,11,12,13,14,15);

#pragma unroll
        for (int n = 0; n < 4; ++n) {
            // B fragment (32x16 f16): lane col = r, K = kk+h*8..+7, kk+16+h*8..+7.
            const _Float16* bp = &sB[n * 16 + r][kk + h * 8];
            v8h blo = *(const v8h*)bp;
            v8h bhi = *(const v8h*)(bp + 16);
            v16h b = __builtin_shufflevector(blo, bhi,
                     0,1,2,3,4,5,6,7,8,9,10,11,12,13,14,15);

            acc[n] = __builtin_amdgcn_wmma_f32_16x16x32_f16(
                false, a, false, b, (short)0, acc[n], false, false);
        }
    }

    // ---- Fused epilogue: C/D layout -> VGPR j holds (M = j + h*8, N = lane&15).
    const float invB = 1.0f / (float)BATCH;
#pragma unroll
    for (int n = 0; n < 4; ++n) {
#pragma unroll
        for (int j = 0; j < 8; ++j) {
            const int row = blockM + wave * 16 + h * 8 + j;
            const int col = blockN + n * 16 + r;
            const float w = weights[row * DIM + col];
            out[row * DIM + col] = LR * (acc[n][j] * invB - WD * w);
        }
    }
}

extern "C" void kernel_launch(void* const* d_in, const int* in_sizes, int n_in,
                              void* d_out, int out_size, void* d_ws, size_t ws_size,
                              hipStream_t stream) {
    (void)in_sizes; (void)n_in; (void)d_ws; (void)ws_size; (void)out_size;
    const float* pre     = (const float*)d_in[0]; // (128, 1024)
    const float* post    = (const float*)d_in[1]; // (128, 1024)
    const float* weights = (const float*)d_in[2]; // (1024, 1024)
    float* out = (float*)d_out;                   // (1024, 1024)

    dim3 grid(DIM / TILE, DIM / TILE);  // 16 x 16 = 256 blocks
    dim3 block(128);                    // 4 wave32s
    hebbian_wmma_kernel<<<grid, block, 0, stream>>>(pre, post, weights, out);
}
